// KalmanVAE_74655121539450
// MI455X (gfx1250) — matrix operand order repaired and checked by reading, not compiled, not run on hardware
//
#include <hip/hip_runtime.h>
#include <hip/hip_bf16.h>

typedef __attribute__((ext_vector_type(16))) _Float16 v16h;
typedef __attribute__((ext_vector_type(8)))  float    v8f;
typedef __attribute__((ext_vector_type(2)))  float    v2f;

#define HIDN 50
#define NG   200   // 4*HIDN
#define ZD   32
#define SD   33    // padded LDS row stride (bank-conflict free on 64 banks)
#define GD   65    // padded Gauss-Jordan row stride
#define AD   8
#define KM   16
#define NTILE 80   // 64 A_t column tiles + 16 C_t column tiles

#if defined(__has_builtin)
#if __has_builtin(__builtin_amdgcn_wmma_f32_16x16x4_f32)
#define HAS_WMMA_F32 1
#endif
#endif
#ifndef HAS_WMMA_F32
#define HAS_WMMA_F32 0
#endif

__device__ __forceinline__ float sigmoidf_(float x) { return 1.0f / (1.0f + __expf(-x)); }

// ---------------------------------------------------------------------------
// Block-cooperative 32x32 matmul on LDS matrices (row stride ld):
//   D = A * (transB ? B^T : B) + (Cinit ? Cinit : 0) + diag*I
// WMMA path: waves 0..3 each own one 16x16 tile; 8 chained K=4 f32 WMMAs.
// Fragment layouts per CDNA5 ISA §7.12.2:
//   f32 A 16x4 : lane row M=l15 (both halves), half selects K {0,1}/{2,3}
//   f32 B 4x16 : lane col N=l15, half selects K {0,1}/{2,3}
//   f32 C 16x16: VGPR r -> (M = r + 8*half, N = l15)
// Caller must __syncthreads() after.
// ---------------------------------------------------------------------------
__device__ __forceinline__ void mm32(float* __restrict__ D,
                                     const float* __restrict__ A,
                                     const float* __restrict__ B,
                                     bool transB,
                                     const float* __restrict__ Cinit,
                                     float diag, int ld, int tid) {
#if HAS_WMMA_F32
  const int wave = tid >> 5;
  if (wave < 4) {
    const int lane = tid & 31;
    const int half = lane >> 4, l15 = lane & 15;
    const int mi = (wave >> 1) << 4;   // 0 or 16
    const int nj = (wave & 1) << 4;    // 0 or 16
    const int col  = nj + l15;
    const int arow = mi + l15;
    v8f acc;
    #pragma unroll
    for (int r = 0; r < 8; ++r) {
      int row = mi + r + 8 * half;
      float c0 = Cinit ? Cinit[row * ld + col] : 0.0f;
      if (row == col) c0 += diag;
      acc[r] = c0;
    }
    #pragma unroll
    for (int kb = 0; kb < 8; ++kb) {
      const int k0 = kb * 4 + 2 * half;
      v2f aa, bb;
      aa[0] = A[arow * ld + k0];
      aa[1] = A[arow * ld + k0 + 1];
      bb[0] = transB ? B[col * ld + k0]     : B[k0 * ld + col];
      bb[1] = transB ? B[col * ld + k0 + 1] : B[(k0 + 1) * ld + col];
      acc = __builtin_amdgcn_wmma_f32_16x16x4_f32(
          /*neg_a=*/false, aa, /*neg_b=*/false, bb,
          /*c_mod=*/(short)0, acc, /*reuse_a=*/false, /*reuse_b=*/false);
    }
    #pragma unroll
    for (int r = 0; r < 8; ++r)
      D[(mi + r + 8 * half) * ld + col] = acc[r];
  }
#else
  for (int e = tid; e < 1024; e += 256) {
    int i = e >> 5, j = e & 31;
    float s = Cinit ? Cinit[i * ld + j] : 0.0f;
    if (i == j) s += diag;
    for (int k = 0; k < ZD; ++k)
      s += A[i * ld + k] * (transB ? B[j * ld + k] : B[k * ld + j]);
    D[i * ld + j] = s;
  }
#endif
}

// ---------------------------------------------------------------------------
// Kernel 1: LSTM over time (one block per batch element) + softmax mixture
// weights written as f16 for the WMMA interpolation GEMM.
// joint[b,0] = a1, joint[b,t] = a[b,t-1]  (t >= 1)
// ---------------------------------------------------------------------------
__global__ void lstm_alpha_kernel(const float* __restrict__ a,
                                  const float* __restrict__ a1,
                                  const float* __restrict__ W_ih,    // [200,8]
                                  const float* __restrict__ W_hh,    // [200,50]
                                  const float* __restrict__ b_ih,
                                  const float* __restrict__ b_hh,
                                  const float* __restrict__ W_alpha, // [16,50]
                                  const float* __restrict__ b_alpha,
                                  _Float16* __restrict__ wh,         // [B*T,16]
                                  int T) {
  const int b   = blockIdx.x;
  const int tid = threadIdx.x;   // 256 threads

  __shared__ float sWhh[NG * HIDN];
  __shared__ float sWih[NG * AD];
  __shared__ float sb[NG];
  __shared__ float sWa[KM * HIDN];
  __shared__ float sba[KM];
  __shared__ float h[HIDN], c[HIDN], x[AD], g[NG], logits[KM];

  for (int e = tid; e < NG * HIDN; e += 256) sWhh[e] = W_hh[e];
  for (int e = tid; e < NG * AD;   e += 256) sWih[e] = W_ih[e];
  for (int e = tid; e < NG;        e += 256) sb[e]   = b_ih[e] + b_hh[e];
  for (int e = tid; e < KM * HIDN; e += 256) sWa[e]  = W_alpha[e];
  if (tid < KM) sba[tid] = b_alpha[tid];
  if (tid < HIDN) { h[tid] = 0.0f; c[tid] = 0.0f; }
  __syncthreads();

  for (int t = 0; t < T; ++t) {
    if (tid < AD)
      x[tid] = (t == 0) ? a1[tid] : a[((size_t)b * T + (t - 1)) * AD + tid];
    __syncthreads();

    if (tid < NG) {
      float s = sb[tid];
      #pragma unroll
      for (int j = 0; j < AD; ++j) s += sWih[tid * AD + j] * x[j];
      for (int j = 0; j < HIDN; ++j) s += sWhh[tid * HIDN + j] * h[j];
      g[tid] = s;
    }
    __syncthreads();

    if (tid < HIDN) {
      float gi = sigmoidf_(g[tid]);
      float gf = sigmoidf_(g[HIDN + tid]);
      float gg = tanhf(g[2 * HIDN + tid]);
      float go = sigmoidf_(g[3 * HIDN + tid]);
      float cn = gf * c[tid] + gi * gg;
      c[tid] = cn;
      h[tid] = go * tanhf(cn);
    }
    __syncthreads();

    if (tid < KM) {
      float s = sba[tid];
      for (int j = 0; j < HIDN; ++j) s += sWa[tid * HIDN + j] * h[j];
      logits[tid] = s;
    }
    __syncthreads();

    if (tid == 0) {
      float mx = logits[0];
      #pragma unroll
      for (int k = 1; k < KM; ++k) mx = fmaxf(mx, logits[k]);
      float sum = 0.0f;
      float e16[KM];
      #pragma unroll
      for (int k = 0; k < KM; ++k) { e16[k] = __expf(logits[k] - mx); sum += e16[k]; }
      float inv = 1.0f / sum;
      _Float16* dst = wh + ((size_t)b * T + t) * KM;
      #pragma unroll
      for (int k = 0; k < KM; ++k) dst[k] = (_Float16)(e16[k] * inv);
    }
    __syncthreads();
  }
}

// ---------------------------------------------------------------------------
// Kernel 2a: one-time staging of the 80 column-tile B fragments in per-lane
// WMMA layout (f16, K padded 16->32 with zeros).  staged[(tn*32+lane)*16+e].
// B fragment layout (16-bit B 32x16): lane col N = l15, half selects K base
// 0/16, element e -> K = kb + e.
// ---------------------------------------------------------------------------
__global__ void stage_bfrag_kernel(const float* __restrict__ Aflat,  // [16,1024]
                                   const float* __restrict__ Cflat,  // [16,256]
                                   _Float16* __restrict__ staged) {
  const int tn   = blockIdx.x;        // 0..79
  const int lane = threadIdx.x;       // 0..31
  const bool isA = (tn < 64);
  const float* src = isA ? Aflat : Cflat;
  const int ldn    = isA ? 1024 : 256;
  const int tileN  = isA ? tn : tn - 64;
  const int col    = tileN * 16 + (lane & 15);
  const int kb     = (lane >> 4) ? 16 : 0;
  _Float16* dst = staged + ((size_t)(tn * 32 + lane) << 4);
  #pragma unroll
  for (int e = 0; e < 16; ++e) {
    int k = kb + e;
    dst[e] = (k < KM) ? (_Float16)src[(size_t)k * ldn + col] : (_Float16)0.0f;
  }
}

// ---------------------------------------------------------------------------
// Kernel 2b: WMMA interpolation GEMM.
//   A_t = w[BT,16] @ Aflat[16,1024],  C_t = w[BT,16] @ Cflat[16,256]
// One wave per block; A-fragment built once (lanes 0-15: 16 contiguous live halfs;
// lanes 16-31 all zero), B fragments fetched as whole v16h vectors from the staged
// table; 8 column tiles fully unrolled -> 8 v_wmma per wave.
// grid = (BT/16, 10): ty 0..7 -> A_t tiles, ty 8..9 -> C_t tiles.
// ---------------------------------------------------------------------------
__global__ void interp_wmma_kernel(const _Float16* __restrict__ wh,
                                   const _Float16* __restrict__ staged,
                                   float* __restrict__ outA,         // [BT,1024]
                                   float* __restrict__ outC) {       // [BT,256]
  const int lane  = threadIdx.x;      // 0..31
  const int tileM = blockIdx.x;       // BT/16 tiles
  const int ty    = blockIdx.y;       // 0..9
  const bool isA  = (ty < 8);
  const int tn0   = ty * 8;           // global tile id base (A: 0..63, C: 64..79)
  float* dst      = isA ? outA : outC;
  const int ldn   = isA ? 1024 : 256;
  const int cn0   = isA ? tn0 * 16 : (tn0 - 64) * 16;  // first output column

  const int half = lane >> 4;         // 0 or 1
  const int l15  = lane & 15;

  // A fragment: 16x32 f16. Lane row M = l15; half 0 holds K in {0..15},
  // half 1 holds K in {16..31}. K >= 16 is compile-time zero padding.
  v16h afrag;
  {
    const _Float16* wrow = wh + ((size_t)tileM * 16 + l15) * KM;
    const int kb = half ? 16 : 0;
    #pragma unroll
    for (int v = 0; v < 8; ++v) {
      #pragma unroll
      for (int h2 = 0; h2 < 2; ++h2) {
        if (kb + 2 * v + h2 < KM) afrag[2 * v + h2] = wrow[kb + 2 * v + h2];      // K < 16
        else       afrag[2 * v + h2] = (_Float16)0.0f;             // K >= 16
      }
    }
  }

  #pragma unroll
  for (int u = 0; u < 8; ++u) {
    const int tn  = tn0 + u;                       // global tile id
    const int col = cn0 + u * 16 + l15;            // output column
    v16h bfrag = *(const v16h*)(staged + ((size_t)(tn * 32 + lane) << 4));
    v8f acc = {};
    acc = __builtin_amdgcn_wmma_f32_16x16x32_f16(
        /*neg_a=*/false, afrag, /*neg_b=*/false, bfrag,
        /*c_mod=*/(short)0, acc, /*reuse_a=*/false, /*reuse_b=*/false);
    #pragma unroll
    for (int r = 0; r < 8; ++r) {
      size_t row = (size_t)tileM * 16 + r + (half ? 8 : 0);
      dst[row * ldn + col] = acc[r];
    }
  }
}

// ---------------------------------------------------------------------------
// Kernel 3: Kalman filter forward scan. One block (256 threads) per batch
// element; 32x32 state in LDS (row stride SD=33, bank-conflict free);
// 8x8 innovation inverse via Gauss-Jordan; 32x32x32 matmuls via f32 WMMA.
// ---------------------------------------------------------------------------
__global__ void kalman_filter_kernel(const float* __restrict__ a,
                                     const float* __restrict__ At,   // [BT,1024]
                                     const float* __restrict__ Ct,   // [BT,256]
                                     float* __restrict__ mu_out,     // [BT,32]
                                     float* __restrict__ sig_out,    // [BT,1024]
                                     float* __restrict__ ws_mu_pred,
                                     float* __restrict__ ws_sig_pred,
                                     float* __restrict__ ws_mu_filt,
                                     float* __restrict__ ws_sig_filt,
                                     int T) {
  const int b = blockIdx.x;
  const int tid = threadIdx.x;

  __shared__ float sigp[ZD * SD], sigf[ZD * SD], tmp[ZD * SD];
  __shared__ float An[ZD * SD], Ac[ZD * SD], KC[ZD * SD];
  __shared__ float Cc[256], CS[256], SCt[256], Kg[256];
  __shared__ float G[8][17];
  __shared__ float fac8[8];
  __shared__ float pivv;
  __shared__ float mu_prev[32], mu_pred[32], Amu[32], muf[32], err[8], y[8];

  for (int e = tid; e < 1024; e += 256) {
    int i = e >> 5, j = e & 31;
    sigp[i * SD + j] = (i == j) ? 20.0f : 0.0f;
  }
  if (tid < 32) { mu_prev[tid] = 0.0f; mu_pred[tid] = 0.0f; }
  __syncthreads();

  for (int t = 0; t < T; ++t) {
    const size_t bt = (size_t)b * T + t;

    for (int e = tid; e < 1024; e += 256)
      Ac[(e >> 5) * SD + (e & 31)] = At[bt * 1024 + e];
    Cc[tid] = Ct[bt * 256 + tid];
    if (tid < AD) y[tid] = a[bt * AD + tid];
    // emit predicted (carry) state for the smoother
    for (int e = tid; e < 1024; e += 256)
      ws_sig_pred[bt * 1024 + e] = sigp[(e >> 5) * SD + (e & 31)];
    if (tid < 32) ws_mu_pred[bt * 32 + tid] = mu_pred[tid];
    if (t + 1 < T) {   // gfx1250 global_prefetch of next step's matrices
      __builtin_prefetch(&At[(bt + 1) * 1024], 0, 1);
      __builtin_prefetch(&Ct[(bt + 1) * 256], 0, 1);
    }
    __syncthreads();

    // CS = C (8x32) * sigp (32x32)
    { int i = tid >> 5, l = tid & 31;
      float s = 0.0f;
      for (int k = 0; k < 32; ++k) s += Cc[i * 32 + k] * sigp[k * SD + l];
      CS[i * 32 + l] = s; }
    __syncthreads();

    // S = CS * C^T + 0.03 I  (augmented for inversion)
    if (tid < 64) {
      int i = tid >> 3, j = tid & 7;
      float s = (i == j) ? 0.03f : 0.0f;
      for (int l = 0; l < 32; ++l) s += CS[i * 32 + l] * Cc[j * 32 + l];
      G[i][j] = s; G[i][8 + j] = (i == j) ? 1.0f : 0.0f;
    }
    __syncthreads();

    // invert S (8x8 Gauss-Jordan, SPD -> no pivoting)
    for (int p = 0; p < 8; ++p) {
      if (tid == 0) pivv = 1.0f / G[p][p];
      __syncthreads();
      if (tid < 16) G[p][tid] *= pivv;
      __syncthreads();
      if (tid < 8 && tid != p) fac8[tid] = G[tid][p];
      __syncthreads();
      if (tid < 128) { int i = tid >> 4, j = tid & 15;
        if (i != p) G[i][j] -= fac8[i] * G[p][j]; }
      __syncthreads();
    }

    // SCt = sigp * C^T (32x8); Kg = SCt * Sinv (32x8)
    { int i = tid >> 3, j = tid & 7;
      float s = 0.0f;
      for (int k = 0; k < 32; ++k) s += sigp[i * SD + k] * Cc[j * 32 + k];
      SCt[i * 8 + j] = s; }
    __syncthreads();
    { int i = tid >> 3, j = tid & 7;
      float s = 0.0f;
      #pragma unroll
      for (int k = 0; k < 8; ++k) s += SCt[i * 8 + k] * G[k][8 + j];
      Kg[i * 8 + j] = s; }
    __syncthreads();

    // Amu = A*mu_prev; err = y - C*Amu; mu_f = Amu + Kg*err
    if (tid < 32) {
      float s = 0.0f;
      for (int k = 0; k < 32; ++k) s += Ac[tid * SD + k] * mu_prev[k];
      Amu[tid] = s;
    }
    __syncthreads();
    if (tid < 8) {
      float s = 0.0f;
      for (int k = 0; k < 32; ++k) s += Cc[tid * 32 + k] * Amu[k];
      err[tid] = y[tid] - s;
    }
    __syncthreads();
    if (tid < 32) {
      float s = Amu[tid];
      #pragma unroll
      for (int k = 0; k < 8; ++k) s += Kg[tid * 8 + k] * err[k];
      muf[tid] = s;
    }
    // KC = Kg (32x8) * C (8x32)
    for (int e = tid; e < 1024; e += 256) {
      int i = e >> 5, j = e & 31;
      float s = 0.0f;
      #pragma unroll
      for (int k = 0; k < 8; ++k) s += Kg[i * 8 + k] * Cc[k * 32 + j];
      KC[i * SD + j] = s;
    }
    __syncthreads();

    // sig_f = sigp - KC * sigp        (WMMA product, then elementwise sub)
    mm32(tmp, KC, sigp, /*transB=*/false, nullptr, 0.0f, SD, tid);
    __syncthreads();
    for (int e = tid; e < 1024; e += 256) {
      int idx = (e >> 5) * SD + (e & 31);
      sigf[idx] = sigp[idx] - tmp[idx];
    }
    __syncthreads();

    for (int e = tid; e < 1024; e += 256)
      ws_sig_filt[bt * 1024 + e] = sigf[(e >> 5) * SD + (e & 31)];
    if (tid < 32) ws_mu_filt[bt * 32 + tid] = muf[tid];
    if (t == T - 1) {  // last smoothed == last filtered
      for (int e = tid; e < 1024; e += 256)
        sig_out[bt * 1024 + e] = sigf[(e >> 5) * SD + (e & 31)];
      if (tid < 32) mu_out[bt * 32 + tid] = muf[tid];
    }

    // A_next = At[min(t+1, T-1)]
    { size_t btn = (size_t)b * T + ((t + 1 < T) ? (t + 1) : (T - 1));
      for (int e = tid; e < 1024; e += 256)
        An[(e >> 5) * SD + (e & 31)] = At[btn * 1024 + e]; }
    __syncthreads();

    // sigp' = An * sigf * An^T + 0.08 I
    mm32(tmp, An, sigf, /*transB=*/false, nullptr, 0.0f, SD, tid);
    __syncthreads();
    mm32(sigp, tmp, An, /*transB=*/true, nullptr, 0.08f, SD, tid);
    // mu_pred' = An * muf ; mu_prev = muf
    if (tid < 32) {
      float s = 0.0f;
      for (int k = 0; k < 32; ++k) s += An[tid * SD + k] * muf[k];
      mu_pred[tid] = s;
    }
    __syncthreads();
    if (tid < 32) mu_prev[tid] = muf[tid];
    __syncthreads();
  }
}

// ---------------------------------------------------------------------------
// Kernel 4: RTS smoother backward scan. One block per batch element; 32x32
// sig_pred inverse via block Gauss-Jordan (stride GD=65); matmuls via WMMA.
// ---------------------------------------------------------------------------
__global__ void rts_smoother_kernel(const float* __restrict__ At,
                                    const float* __restrict__ ws_mu_pred,
                                    const float* __restrict__ ws_sig_pred,
                                    const float* __restrict__ ws_mu_filt,
                                    const float* __restrict__ ws_sig_filt,
                                    float* __restrict__ mu_out,
                                    float* __restrict__ sig_out,
                                    int T) {
  const int b = blockIdx.x;
  const int tid = threadIdx.x;

  __shared__ float G[32 * GD];
  __shared__ float An[ZD * SD], sf[ZD * SD], sp[ZD * SD], spinv[ZD * SD];
  __shared__ float J[ZD * SD], T1[ZD * SD], T2[ZD * SD], Dm[ZD * SD];
  __shared__ float sig_sn[ZD * SD];
  __shared__ float mf[32], mp[32], mu_sn[32], dvec[32], fac[32];
  __shared__ float pivv;

  const size_t btl = (size_t)b * T + (T - 1);
  for (int e = tid; e < 1024; e += 256)
    sig_sn[(e >> 5) * SD + (e & 31)] = ws_sig_filt[btl * 1024 + e];
  if (tid < 32) mu_sn[tid] = ws_mu_filt[btl * 32 + tid];
  __syncthreads();

  for (int t = T - 2; t >= 0; --t) {
    const size_t bt  = (size_t)b * T + t;
    const size_t btn = bt + 1;
    for (int e = tid; e < 1024; e += 256) {
      int idx = (e >> 5) * SD + (e & 31);
      sf[idx] = ws_sig_filt[bt * 1024 + e];
      sp[idx] = ws_sig_pred[btn * 1024 + e];
      An[idx] = At[btn * 1024 + e];
    }
    if (tid < 32) { mf[tid] = ws_mu_filt[bt * 32 + tid];
                    mp[tid] = ws_mu_pred[btn * 32 + tid]; }
    if (t > 0) __builtin_prefetch(&ws_sig_pred[bt * 1024], 0, 1);
    __syncthreads();

    // invert sp: augmented [sp | I], Gauss-Jordan (SPD -> no pivoting)
    for (int e = tid; e < 2048; e += 256) {
      int i = e >> 6, j = e & 63;
      G[i * GD + j] = (j < 32) ? sp[i * SD + j] : (((j - 32) == i) ? 1.0f : 0.0f);
    }
    __syncthreads();
    for (int p = 0; p < 32; ++p) {
      if (tid == 0) pivv = 1.0f / G[p * GD + p];
      __syncthreads();
      if (tid < 64) G[p * GD + tid] *= pivv;
      __syncthreads();
      if (tid < 32 && tid != p) fac[tid] = G[tid * GD + p];
      __syncthreads();
      for (int e = tid; e < 2048; e += 256) {
        int i = e >> 6, j = e & 63;
        if (i != p) G[i * GD + j] -= fac[i] * G[p * GD + j];
      }
      __syncthreads();
    }
    for (int e = tid; e < 1024; e += 256) {
      int i = e >> 5, j = e & 31;
      spinv[i * SD + j] = G[i * GD + 32 + j];
    }
    __syncthreads();

    // J = sf * An^T * spinv
    mm32(T1, sf, An, /*transB=*/true, nullptr, 0.0f, SD, tid);
    __syncthreads();
    mm32(J, T1, spinv, /*transB=*/false, nullptr, 0.0f, SD, tid);
    // differences (independent of J; all waves)
    if (tid < 32) dvec[tid] = mu_sn[tid] - mp[tid];
    for (int e = tid; e < 1024; e += 256) {
      int idx = (e >> 5) * SD + (e & 31);
      Dm[idx] = sig_sn[idx] - sp[idx];
    }
    __syncthreads();

    // mu_s = mf + J * dvec ; T2 = J * Dm
    if (tid < 32) {
      float s = mf[tid];
      for (int k = 0; k < 32; ++k) s += J[tid * SD + k] * dvec[k];
      mu_sn[tid] = s;
    }
    mm32(T2, J, Dm, /*transB=*/false, nullptr, 0.0f, SD, tid);
    __syncthreads();
    // sig_s = sf + T2 * J^T  (overwrites carry)
    mm32(sig_sn, T2, J, /*transB=*/true, sf, 0.0f, SD, tid);
    __syncthreads();

    for (int e = tid; e < 1024; e += 256)
      sig_out[bt * 1024 + e] = sig_sn[(e >> 5) * SD + (e & 31)];
    if (tid < 32) mu_out[bt * 32 + tid] = mu_sn[tid];
    __syncthreads();
  }
}

// ---------------------------------------------------------------------------
extern "C" void kernel_launch(void* const* d_in, const int* in_sizes, int n_in,
                              void* d_out, int out_size, void* d_ws, size_t ws_size,
                              hipStream_t stream) {
  const float* a       = (const float*)d_in[0];  // [256,128,8]
  const float* a1      = (const float*)d_in[1];  // [8]
  const float* W_ih    = (const float*)d_in[2];  // [200,8]
  const float* W_hh    = (const float*)d_in[3];  // [200,50]
  const float* b_ih    = (const float*)d_in[4];  // [200]
  const float* b_hh    = (const float*)d_in[5];  // [200]
  const float* W_alpha = (const float*)d_in[6];  // [16,50]
  const float* b_alpha = (const float*)d_in[7];  // [16]
  const float* A       = (const float*)d_in[8];  // [16,32,32] == [16,1024]
  const float* C       = (const float*)d_in[9];  // [16,8,32]  == [16,256]

  const int B = 256, T = 128;
  const size_t BT = (size_t)B * T;

  float* out     = (float*)d_out;
  float* mu_out  = out;                         // [BT,32]
  float* sig_out = mu_out + BT * 32;            // [BT,1024]
  float* A_out   = sig_out + BT * 1024;         // [BT,1024]
  float* C_out   = A_out + BT * 1024;           // [BT,256]

  char* ws = (char*)d_ws;
  _Float16* wh = (_Float16*)ws;              ws += BT * KM * sizeof(_Float16);
  _Float16* stagedB = (_Float16*)ws;         ws += (size_t)NTILE * 32 * 16 * sizeof(_Float16);
  float* ws_mu_pred  = (float*)ws;           ws += BT * 32   * sizeof(float);
  float* ws_mu_filt  = (float*)ws;           ws += BT * 32   * sizeof(float);
  float* ws_sig_pred = (float*)ws;           ws += BT * 1024 * sizeof(float);
  float* ws_sig_filt = (float*)ws;           ws += BT * 1024 * sizeof(float);

  // 1) LSTM + mixture weights (sequential in T, parallel over B)
  lstm_alpha_kernel<<<B, 256, 0, stream>>>(a, a1, W_ih, W_hh, b_ih, b_hh,
                                           W_alpha, b_alpha, wh, T);

  // 2a) stage B fragments (padded, per-lane WMMA layout)
  stage_bfrag_kernel<<<NTILE, 32, 0, stream>>>(A, C, stagedB);

  // 2b) WMMA interpolation GEMM: 8 column tiles per wave (64 A + 16 C tiles)
  dim3 ggemm((unsigned)(BT / 16), 10);
  interp_wmma_kernel<<<ggemm, 32, 0, stream>>>(wh, stagedB, A_out, C_out);

  // 3) Kalman filter forward scan
  kalman_filter_kernel<<<B, 256, 0, stream>>>(a, A_out, C_out, mu_out, sig_out,
                                              ws_mu_pred, ws_sig_pred,
                                              ws_mu_filt, ws_sig_filt, T);

  // 4) RTS smoother backward scan
  rts_smoother_kernel<<<B, 256, 0, stream>>>(A_out, ws_mu_pred, ws_sig_pred,
                                             ws_mu_filt, ws_sig_filt,
                                             mu_out, sig_out, T);
}
